// MambaBlock_9062380995099
// MI455X (gfx1250) — matrix operand order repaired and checked by reading, not compiled, not run on hardware
//
#include <hip/hip_runtime.h>
#include <math.h>
#include <stdint.h>

typedef __attribute__((ext_vector_type(16))) _Float16 v16h;
typedef __attribute__((ext_vector_type(8)))  _Float16 v8h;
typedef __attribute__((ext_vector_type(8)))  float    v8f;

#define DIM      1024
#define D_INNER  2048
#define BATCH    4
#define SEQ      2048
#define BL       (BATCH * SEQ)      // 8192 rows in every GEMM
#define BPAD     40                 // halfs per LDS B row: 32 data + 8 pad (80B, 16B aligned)

enum { EPI_NONE = 0, EPI_SOFTPLUS = 1, EPI_EXP = 2, EPI_MUL1 = 3, EPI_MUL2 = 4 };

// ---------------------------------------------------------------------------
// WMMA fragment loaders (wave32, v_wmma_f32_16x16x32_f16, ISA 7.12.2)
// A 16x32: lane holds row M=lane%16, K runs [kb,kb+8) and [kb+16,kb+24), kb=(lane/16)*8
// B 32x16: lane holds col N=lane%16, K runs [kb,kb+8) and [kb+16,kb+24), kb=(lane/16)*8
// ---------------------------------------------------------------------------
__device__ __forceinline__ v16h frag_A_f16(const _Float16* __restrict__ base,
                                           int ld, int lane) {
  const int r  = lane & 15;
  const int kb = (lane >> 4) << 3;
  const _Float16* p = base + (size_t)r * ld + kb;
  v8h lo = *(const v8h*)p;
  v8h hi = *(const v8h*)(p + 16);
  v16h f;
#pragma unroll
  for (int i = 0; i < 8; ++i) { f[i] = lo[i]; f[8 + i] = hi[i]; }
  return f;
}

__device__ __forceinline__ v16h frag_B_lds(const _Float16* rowp, int lane) {
  const _Float16* p = rowp + ((lane >> 4) << 3);
  v8h lo = *(const v8h*)p;
  v8h hi = *(const v8h*)(p + 16);
  v16h f;
#pragma unroll
  for (int i = 0; i < 8; ++i) { f[i] = lo[i]; f[8 + i] = hi[i]; }
  return f;
}

// Stage one 128(N) x 32(K) f16 weight tile into LDS via async-to-LDS DMA.
// 512 16-byte chunks distributed over 256 threads (2 per thread), ASYNCcnt-tracked.
__device__ __forceinline__ void stage_B(const _Float16* __restrict__ B, int ldb,
                                        int n_blk, int k0, _Float16* bufp) {
  const int t = threadIdx.x;
#pragma unroll
  for (int c = 0; c < 2; ++c) {
    const int e = t + c * 256;
    const int row = e >> 2;
    const int q = e & 3;
    const _Float16* gp = B + (size_t)(n_blk + row) * ldb + k0 + q * 8;
    uint32_t lds = (uint32_t)(uintptr_t)(bufp + row * BPAD + q * 8);
    unsigned long long ga = (unsigned long long)(uintptr_t)gp;
    asm volatile("global_load_async_to_lds_b128 %0, %1, off"
                 :: "v"(lds), "v"(ga) : "memory");
  }
}

template <int MODE>
__device__ __forceinline__ void emit(float accv, int row, int col,
                                     float* __restrict__ Cf,
                                     _Float16* __restrict__ Ch, int ldc,
                                     const float* __restrict__ E1, int lde1,
                                     const float* __restrict__ E2, int lde2) {
  float v = accv;
  if (MODE == EPI_SOFTPLUS)      v = (v > 20.0f) ? v : log1pf(__expf(v));
  else if (MODE == EPI_EXP)      v = __expf(v);
  else if (MODE == EPI_MUL1)     v *= E1[(size_t)row * lde1 + col];
  else if (MODE == EPI_MUL2)     v *= E1[(size_t)row * lde1 + col] *
                                      E2[(size_t)row * lde2 + col];
  if (Cf) Cf[(size_t)row * ldc + col] = v;
  if (Ch) Ch[(size_t)row * ldc + col] = (_Float16)v;
}

// C[M,N] = epi( A[M,K](f16) @ W[N,K]^T(f16) )
// block tile 128x128 (8 wave32s as 4Mx2N), wave tile 32x64: 8 WMMAs / K-step.
// B tile double-buffered in LDS via async DMA; A streams direct (f16, b128 loads).
template <int MODE>
__global__ __launch_bounds__(256) void gemm_f16_wmma(
    const _Float16* __restrict__ A, int lda,
    const _Float16* __restrict__ B, int ldb,
    float* __restrict__ Cf, _Float16* __restrict__ Ch, int ldc,
    int M, int N, int K,
    const float* __restrict__ E1, int lde1,
    const float* __restrict__ E2, int lde2) {
  __shared__ _Float16 Bs[2][128 * BPAD];

  const int lane = threadIdx.x & 31;
  const int wave = threadIdx.x >> 5;
  const int wm = wave & 3;                    // 4 waves along M
  const int wn = wave >> 2;                   // 2 waves along N
  const int m0 = blockIdx.y * 128 + wm * 32;
  const int n_blk = blockIdx.x * 128;
  const int nw = wn * 64;                     // wave's N offset inside block tile

  v8f acc[2][4] = {};

  stage_B(B, ldb, n_blk, 0, &Bs[0][0]);

  for (int k0 = 0; k0 < K; k0 += 32) {
    const int buf = (k0 >> 5) & 1;
    // my async ops for buffer `buf` complete ...
    asm volatile("s_wait_asynccnt 0" ::: "memory");
    // ... then everyone's have (barrier), and all reads of buf^1 are closed,
    // so it is safe to start the next DMA into buf^1 and overlap with compute.
    __syncthreads();
    if (k0 + 32 < K) stage_B(B, ldb, n_blk, k0 + 32, &Bs[buf ^ 1][0]);

    const _Float16* Abase = A + (size_t)m0 * lda + k0;
    v16h a0 = frag_A_f16(Abase, lda, lane);
    v16h a1 = frag_A_f16(Abase + (size_t)16 * lda, lda, lane);

#pragma unroll
    for (int f = 0; f < 4; ++f) {
      v16h b = frag_B_lds(&Bs[buf][(nw + f * 16 + (lane & 15)) * BPAD], lane);
      acc[0][f] = __builtin_amdgcn_wmma_f32_16x16x32_f16(false, a0, false, b,
                                                         (short)0, acc[0][f], false, false);
      acc[1][f] = __builtin_amdgcn_wmma_f32_16x16x32_f16(false, a1, false, b,
                                                         (short)0, acc[1][f], false, false);
    }
  }

  // C/D layout: col = lane%16 ; VGPR i -> row = i + (lane/16)*8
  const int col0 = n_blk + nw + (lane & 15);
  const int rb = m0 + ((lane >> 4) << 3);
#pragma unroll
  for (int f = 0; f < 4; ++f) {
    const int col = col0 + f * 16;
#pragma unroll
    for (int i = 0; i < 8; ++i) {
      emit<MODE>(acc[0][f][i], rb + i,      col, Cf, Ch, ldc, E1, lde1, E2, lde2);
      emit<MODE>(acc[1][f][i], rb + 16 + i, col, Cf, Ch, ldc, E1, lde1, E2, lde2);
    }
  }
}

// ---------------------------------------------------------------------------
__global__ __launch_bounds__(256) void cvt_f16_kernel(const float* __restrict__ in,
                                                      _Float16* __restrict__ out,
                                                      size_t n) {
  size_t i = (size_t)blockIdx.x * blockDim.x + threadIdx.x;
  if (i < n) out[i] = (_Float16)in[i];
}

__global__ __launch_bounds__(256) void cvt_negexp_f16_kernel(const float* __restrict__ in,
                                                             _Float16* __restrict__ out,
                                                             size_t n) {
  size_t i = (size_t)blockIdx.x * blockDim.x + threadIdx.x;
  if (i < n) out[i] = (_Float16)(-__expf(in[i]));
}

// depthwise conv3 ('same') + bias + silu; writes f32 (scan/epilogue) and f16 (GEMM A)
__global__ __launch_bounds__(256) void conv_silu_kernel(
    const float* __restrict__ xz, const float* __restrict__ cw,
    const float* __restrict__ cb, float* __restrict__ u32o,
    _Float16* __restrict__ u16o) {
  size_t i = (size_t)blockIdx.x * blockDim.x + threadIdx.x;
  if (i >= (size_t)BL * D_INNER) return;
  const int    d  = (int)(i & (D_INNER - 1));
  const size_t bl = i >> 11;
  const int    l  = (int)(bl & (SEQ - 1));
  const float w0 = cw[3 * d], w1 = cw[3 * d + 1], w2 = cw[3 * d + 2];
  float acc = cb[d] + w1 * xz[bl * (size_t)(2 * D_INNER) + d];
  if (l > 0)       acc += w0 * xz[(bl - 1) * (size_t)(2 * D_INNER) + d];
  if (l < SEQ - 1) acc += w2 * xz[(bl + 1) * (size_t)(2 * D_INNER) + d];
  const float s = acc / (1.0f + __expf(-acc));
  u32o[i] = s;
  u16o[i] = (_Float16)s;
}

// selective scan: thread per (b,d); d-contiguous across lanes -> coalesced
__global__ __launch_bounds__(256) void scan_kernel(
    const float* __restrict__ dA, const float* __restrict__ dBu,
    const float* __restrict__ u, _Float16* __restrict__ xs16_out,
    float* __restrict__ ds_out) {
  const int t = blockIdx.x * blockDim.x + threadIdx.x;
  if (t >= BATCH * D_INNER) return;
  const int d = t & (D_INNER - 1);
  const int b = t >> 11;
  size_t idx = (size_t)b * SEQ * D_INNER + d;
  float xs = 0.0f, ds = 1.0f;
  for (int l = 0; l < SEQ; ++l, idx += D_INNER) {
    xs = fmaf(dA[idx], xs, dBu[idx]);
    ds = fmaf(u[idx], xs, ds);
    xs16_out[idx] = (_Float16)xs;
    ds_out[idx] = ds;
  }
}

// ---------------------------------------------------------------------------
extern "C" void kernel_launch(void* const* d_in, const int* in_sizes, int n_in,
                              void* d_out, int out_size, void* d_ws, size_t ws_size,
                              hipStream_t stream) {
  const float* x      = (const float*)d_in[0];
  const float* W_in   = (const float*)d_in[1];
  const float* conv_w = (const float*)d_in[2];
  const float* conv_b = (const float*)d_in[3];
  const float* W_dt   = (const float*)d_in[4];
  const float* A_proj = (const float*)d_in[5];
  const float* B_proj = (const float*)d_in[6];
  const float* C_proj = (const float*)d_in[7];
  const float* W_out  = (const float*)d_in[8];
  float* out = (float*)d_out;

  char* ws = (char*)d_ws;
  float*    xz      = (float*)(ws + 0ull);            // [8192,4096] u_raw | z
  float*    u32     = (float*)(ws + 134217728ull);    // [8192,2048]
  float*    dA      = (float*)(ws + 201326592ull);    // [8192,2048]
  float*    dBu     = (float*)(ws + 268435456ull);    // [8192,2048]
  float*    dsb     = (float*)(ws + 335544320ull);    // [8192,2048]
  _Float16* x16     = (_Float16*)(ws + 402653184ull); // [8192,1024]
  _Float16* u16     = (_Float16*)(ws + 419430400ull); // [8192,2048]
  _Float16* delta16 = (_Float16*)(ws + 452984832ull); // [8192,2048]
  _Float16* y16     = delta16;                        // delta dead after GEMM 5
  _Float16* xs16    = (_Float16*)(ws + 486539264ull); // [8192,2048]
  _Float16* W_in16  = (_Float16*)(ws + 520093696ull);
  _Float16* W_dt16  = W_in16 + 4096ull * 1024;
  _Float16* A16     = W_dt16 + 2048ull * 2048;
  _Float16* B16     = A16    + 2048ull * 2048;
  _Float16* C16     = B16    + 2048ull * 2048;
  _Float16* W_out16 = C16    + 2048ull * 2048;

  const dim3 blk(256);
  const size_t n8M = 8388608, n4M = 4194304, n2M = 2097152;
  cvt_f16_kernel<<<(n8M + 255) / 256, blk, 0, stream>>>(x, x16, n8M);
  cvt_f16_kernel<<<(n4M + 255) / 256, blk, 0, stream>>>(W_in, W_in16, n4M);
  cvt_f16_kernel<<<(n4M + 255) / 256, blk, 0, stream>>>(W_dt, W_dt16, n4M);
  cvt_negexp_f16_kernel<<<(n4M + 255) / 256, blk, 0, stream>>>(A_proj, A16, n4M);
  cvt_f16_kernel<<<(n4M + 255) / 256, blk, 0, stream>>>(B_proj, B16, n4M);
  cvt_f16_kernel<<<(n4M + 255) / 256, blk, 0, stream>>>(C_proj, C16, n4M);
  cvt_f16_kernel<<<(n2M + 255) / 256, blk, 0, stream>>>(W_out, W_out16, n2M);

  // 1) xz = x @ W_in^T                         [8192,4096] f32
  gemm_f16_wmma<EPI_NONE><<<dim3(4096 / 128, BL / 128), blk, 0, stream>>>(
      x16, DIM, W_in16, DIM, xz, nullptr, 2 * D_INNER, BL, 2 * D_INNER, DIM,
      nullptr, 0, nullptr, 0);

  // 2) u = silu(conv3(u_raw) + b)              f32 + f16
  conv_silu_kernel<<<((size_t)BL * D_INNER + 255) / 256, blk, 0, stream>>>(
      xz, conv_w, conv_b, u32, u16);

  // 3) delta = softplus(u @ W_dt^T)            f16 only (GEMM-A input only)
  gemm_f16_wmma<EPI_SOFTPLUS><<<dim3(D_INNER / 128, BL / 128), blk, 0, stream>>>(
      u16, D_INNER, W_dt16, D_INNER, nullptr, delta16, D_INNER,
      BL, D_INNER, D_INNER, nullptr, 0, nullptr, 0);

  // 4) dA = exp(delta @ A^T)                   f32 (scan input)
  gemm_f16_wmma<EPI_EXP><<<dim3(D_INNER / 128, BL / 128), blk, 0, stream>>>(
      delta16, D_INNER, A16, D_INNER, dA, nullptr, D_INNER,
      BL, D_INNER, D_INNER, nullptr, 0, nullptr, 0);

  // 5) dBu = (delta @ B^T) * u                 f32 (scan input)
  gemm_f16_wmma<EPI_MUL1><<<dim3(D_INNER / 128, BL / 128), blk, 0, stream>>>(
      delta16, D_INNER, B16, D_INNER, dBu, nullptr, D_INNER,
      BL, D_INNER, D_INNER, u32, D_INNER, nullptr, 0);

  // 6) selective scan over SEQ -> xs (f16), ds (f32)
  scan_kernel<<<(BATCH * D_INNER) / 256, blk, 0, stream>>>(dA, dBu, u32, xs16, dsb);

  // 7) y = (xs @ C^T) * z * ds                 f16 only (z = xz cols [2048,4096))
  gemm_f16_wmma<EPI_MUL2><<<dim3(D_INNER / 128, BL / 128), blk, 0, stream>>>(
      xs16, D_INNER, C16, D_INNER, nullptr, y16, D_INNER,
      BL, D_INNER, D_INNER, xz + D_INNER, 2 * D_INNER, dsb, D_INNER);

  // 8) out = y @ W_out^T                       f32 -> d_out
  gemm_f16_wmma<EPI_NONE><<<dim3(DIM / 128, BL / 128), blk, 0, stream>>>(
      y16, D_INNER, W_out16, D_INNER, out, nullptr, DIM,
      BL, DIM, D_INNER, nullptr, 0, nullptr, 0);
}
